// SpatialBlockConv2d_45543833207249
// MI455X (gfx1250) — compile-verified, matrix operand order checked by reading, not run on hardware
//
#include <hip/hip_runtime.h>
#include <hip/hip_bf16.h>

typedef __attribute__((ext_vector_type(16))) _Float16 v16h;
typedef __attribute__((ext_vector_type(8)))  _Float16 v8h;
typedef __attribute__((ext_vector_type(8)))  float    v8f;

// Problem constants
#define NB   512   // batch
#define NC   256   // channels
#define NH   16    // H = W
#define NPOS 256   // H*W
#define NG   16    // 4x4 spatial blocks
#define NTAP 9     // 3x3 taps

// Workspace layout (f16):
//   Wh: [g][tap][co][ci] = 16*9*256*256 halves = 18,874,368 bytes
//   Xh: [pos][b][ci]     = 256*512*256 halves  = 67,108,864 bytes
#define WH_ELEMS  (NG * NTAP * NC * NC)
#define XH_OFFSET ((size_t)WH_ELEMS * 2)

// ---------------------------------------------------------------------------
// Kernel 1: W [g][co][ci][kh][kw] f32 -> Wh [g][tap][co][ci] f16
// ---------------------------------------------------------------------------
__global__ __launch_bounds__(256) void sbconv_wprep(const float* __restrict__ W,
                                                    _Float16* __restrict__ Wh) {
    int idx = blockIdx.x * 256 + threadIdx.x;           // over WH_ELEMS
    int ci  = idx & 255;
    int t1  = idx >> 8;
    int co  = t1 & 255;
    int t2  = t1 >> 8;                                  // g*9 + tap
    int tap = t2 % 9;
    int g   = t2 / 9;
    size_t src = ((((size_t)g * NC + co) * NC + ci) * NTAP) + tap;
    Wh[idx] = (_Float16)W[src];
}

// ---------------------------------------------------------------------------
// Kernel 2: x [b][ci][pos] f32 -> Xh [pos][b][ci] f16   (LDS tile transpose)
// block (32,8), grid (8, 8, 512)
// ---------------------------------------------------------------------------
__global__ __launch_bounds__(256) void sbconv_xprep(const float* __restrict__ x,
                                                    _Float16* __restrict__ Xh) {
    __shared__ float tile[32][33];
    int ci0  = blockIdx.x * 32;
    int pos0 = blockIdx.y * 32;
    int b    = blockIdx.z;
    int tx = threadIdx.x, ty = threadIdx.y;
#pragma unroll
    for (int j = 0; j < 4; ++j) {
        int ci  = ci0 + ty + j * 8;
        int pos = pos0 + tx;
        tile[ty + j * 8][tx] = x[((size_t)b * NC + ci) * NPOS + pos];
    }
    __syncthreads();
#pragma unroll
    for (int j = 0; j < 4; ++j) {
        int pos = pos0 + ty + j * 8;
        int ci  = ci0 + tx;
        Xh[((size_t)pos * NB + b) * NC + ci] = (_Float16)tile[tx][ty + j * 8];
    }
}

// ---------------------------------------------------------------------------
// Kernel 3: implicit-GEMM block conv via v_wmma_f32_16x16x32_f16.
// WG = (g, y, 64-co tile, 128-batch tile); 8 waves 2(M)x4(N), wave = 32x32
// (4 WMMAs / 8 fragment ds_loads per k-step). LDS double-buffered: one
// barrier per k-step, next step's global->LDS staging overlaps WMMA.
// Tap bases tracked incrementally (no div/mod in the pipelined loop).
// Staging pitch 40 halves (80B) -> conflict-free ds_load_b128 fragments.
// ---------------------------------------------------------------------------
#define LPITCH 40
#define ATILE  (64 * LPITCH)
#define BTILE  (128 * LPITCH)

__global__ __launch_bounds__(256) void sbconv_gemm(const _Float16* __restrict__ Wh,
                                                   const _Float16* __restrict__ Xh,
                                                   float* __restrict__ out) {
    __shared__ _Float16 sA[2 * ATILE];
    __shared__ _Float16 sB[2 * BTILE];

    int id = blockIdx.x;
    int nt = id & 3;  id >>= 2;      // batch tile   0..3
    int mt = id & 3;  id >>= 2;      // co tile      0..3
    int y  = id & 3;  id >>= 2;      // row in block 0..3
    int g  = id;                     // block        0..15
    int gy = g >> 2, gx = g & 3;
    int m0 = mt * 64, n0 = nt * 128;
    int h  = gy * 4 + y;

    int tid  = threadIdx.x;
    int lane = tid & 31;
    int wv   = tid >> 5;
    int wm   = wv & 1;               // wave M position (0..1) -> co base wm*32
    int wn   = wv >> 1;              // wave N position (0..3) -> b  base wn*32
    int l16  = lane & 15;
    int hi   = (lane & 16) ? 1 : 0;
    int koff = hi ? 8 : 0;           // K-halves offset per 16-bit WMMA layout

    int srow = tid >> 2;             // staging row (0..63)
    int scc  = tid & 3;              // 8-half chunk within 32-K slice

    // Per-thread-constant staging offsets (elements)
    const int aoff  = (srow) * NC + scc * 8;        // + m0*NC folded into Ag row
    const int boff0 = srow * NC + scc * 8;
    const int boff1 = boff0 + 64 * NC;
    const int ldsoA = srow * LPITCH + scc * 8;
    const int ldsoB0 = ldsoA;
    const int ldsoB1 = (64 + srow) * LPITCH + scc * 8;

    int khlo = (y == 0) ? 1 : 0;
    int khhi = (y == 3) ? 1 : 2;
    int nh   = khhi - khlo + 1;

    // A rows for this WG's co tile start at m0
    const _Float16* Ag = Wh + (size_t)g * NTAP * NC * NC + (size_t)m0 * NC;

    for (int xq = 0; xq < 4; ++xq) {
        int w    = gx * 4 + xq;
        int kwlo = (xq == 0) ? 1 : 0;
        int kwhi = (xq == 3) ? 1 : 2;
        int nw   = kwhi - kwlo + 1;
        int S    = nh * nw * 8;      // pipelined steps: taps x (256/32) k-chunks

        v8f acc00 = {}, acc01 = {}, acc10 = {}, acc11 = {};

        // Incrementally tracked tap state for the NEXT stage to issue
        int kh = khlo, kw = kwlo;
        const _Float16* Ab = Ag + (size_t)(kh * 3 + kw) * NC * NC;
        const _Float16* Bb = Xh +
            ((size_t)((h + kh - 1) * NH + (w + kw - 1)) * NB + n0) * NC;

        auto stage = [&](const _Float16* Ap, const _Float16* Bp, int k0, int bufi) {
            _Float16* dA = sA + bufi * ATILE;
            _Float16* dB = sB + bufi * BTILE;
            *(uint4*)(dA + ldsoA)  = *(const uint4*)(Ap + k0 + aoff);
            *(uint4*)(dB + ldsoB0) = *(const uint4*)(Bp + k0 + boff0);
            *(uint4*)(dB + ldsoB1) = *(const uint4*)(Bp + k0 + boff1);
        };

        __syncthreads();             // previous x's final fragment reads done
        stage(Ab, Bb, 0, 0);

        for (int s = 0; s < S; ++s) {
            int cur = s & 1;
            __syncthreads();         // staging of step s complete
            if (s + 1 < S) {
                int k0 = ((s + 1) & 7) << 5;
                if (k0 == 0) {       // advance tap once per 8 steps (uniform)
                    ++kw;
                    if (kw > kwhi) { kw = kwlo; ++kh; }
                    Ab = Ag + (size_t)(kh * 3 + kw) * NC * NC;
                    Bb = Xh + ((size_t)((h + kh - 1) * NH + (w + kw - 1)) * NB + n0) * NC;
                }
                stage(Ab, Bb, k0, cur ^ 1);   // overlaps with WMMAs below
            }

            const _Float16* pA = sA + cur * ATILE;
            const _Float16* pB = sB + cur * BTILE;

            int rA0 = (wm * 32 + l16) * LPITCH;
            int rA1 = (wm * 32 + 16 + l16) * LPITCH;
            v8h a00 = *(const v8h*)(pA + rA0 + koff);
            v8h a01 = *(const v8h*)(pA + rA0 + 16 + koff);
            v8h a10 = *(const v8h*)(pA + rA1 + koff);
            v8h a11 = *(const v8h*)(pA + rA1 + 16 + koff);

            int rB0 = (wn * 32 + l16) * LPITCH;
            int rB1 = (wn * 32 + 16 + l16) * LPITCH;
            v8h b00 = *(const v8h*)(pB + rB0 + koff);
            v8h b01 = *(const v8h*)(pB + rB0 + 16 + koff);
            v8h b10 = *(const v8h*)(pB + rB1 + koff);
            v8h b11 = *(const v8h*)(pB + rB1 + 16 + koff);

            v16h af0 = __builtin_shufflevector(a00, a01, 0,1,2,3,4,5,6,7,8,9,10,11,12,13,14,15);
            v16h af1 = __builtin_shufflevector(a10, a11, 0,1,2,3,4,5,6,7,8,9,10,11,12,13,14,15);
            v16h bf0 = __builtin_shufflevector(b00, b01, 0,1,2,3,4,5,6,7,8,9,10,11,12,13,14,15);
            v16h bf1 = __builtin_shufflevector(b10, b11, 0,1,2,3,4,5,6,7,8,9,10,11,12,13,14,15);

            acc00 = __builtin_amdgcn_wmma_f32_16x16x32_f16(
                false, af0, false, bf0, (short)0, acc00, false, false);
            acc01 = __builtin_amdgcn_wmma_f32_16x16x32_f16(
                false, af0, false, bf1, (short)0, acc01, false, false);
            acc10 = __builtin_amdgcn_wmma_f32_16x16x32_f16(
                false, af1, false, bf0, (short)0, acc10, false, false);
            acc11 = __builtin_amdgcn_wmma_f32_16x16x32_f16(
                false, af1, false, bf1, (short)0, acc11, false, false);
        }

        // Write back. C tile layout: VGPR r -> M = r (lanes<16) / 8+r (lanes>=16),
        // N = lane%16.  M = co, N = batch.
        size_t posidx = (size_t)h * NH + w;
        int bA = n0 + wn * 32 + l16;             // n-frag 0
        int bB = bA + 16;                        // n-frag 1
        int coA = m0 + wm * 32 + (hi ? 8 : 0);   // m-frag 0
        int coB = coA + 16;                      // m-frag 1
#pragma unroll
        for (int r = 0; r < 8; ++r) {
            out[(((size_t)bA * NC + coA + r) << 8) + posidx] = acc00[r];
            out[(((size_t)bB * NC + coA + r) << 8) + posidx] = acc01[r];
            out[(((size_t)bA * NC + coB + r) << 8) + posidx] = acc10[r];
            out[(((size_t)bB * NC + coB + r) << 8) + posidx] = acc11[r];
        }
    }
}

// ---------------------------------------------------------------------------
extern "C" void kernel_launch(void* const* d_in, const int* in_sizes, int n_in,
                              void* d_out, int out_size, void* d_ws, size_t ws_size,
                              hipStream_t stream) {
    const float* x = (const float*)d_in[0];   // [512,256,16,16]
    const float* W = (const float*)d_in[1];   // [16,256,256,3,3]
    float* out = (float*)d_out;               // [512,256,16,16]

    _Float16* Wh = (_Float16*)d_ws;
    _Float16* Xh = (_Float16*)((char*)d_ws + XH_OFFSET);

    sbconv_wprep<<<WH_ELEMS / 256, 256, 0, stream>>>(W, Wh);
    sbconv_xprep<<<dim3(8, 8, NB), dim3(32, 8), 0, stream>>>(x, Xh);
    // 16 g * 4 y * 4 co-tiles * 4 b-tiles = 1024 WGs
    sbconv_gemm<<<1024, 256, 0, stream>>>(Wh, Xh, out);
}
